// ResidualBlock_70884140253235
// MI455X (gfx1250) — matrix operand order repaired and checked by reading, not compiled, not run on hardware
//
#include <hip/hip_runtime.h>
#include <hip/hip_bf16.h>
#include <stdint.h>

// ---------------------------------------------------------------------------
// CDNA5 (gfx1250) implementation of the sparse-conv residual block.
// GEMMs via v_wmma_f32_16x16x32_f16; weight staging via async-to-LDS DMA.
// ---------------------------------------------------------------------------

typedef _Float16 half_t;
typedef __attribute__((ext_vector_type(16))) _Float16 v16h;
typedef __attribute__((ext_vector_type(8)))  _Float16 v8h;
typedef __attribute__((ext_vector_type(8)))  float    v8f;

union V16 { v16h v; v8h h[2]; };

static constexpr int KOFF = 27;   // 3^3 kernel offsets

// ---------------------------------------------------------------------------
// BN statistics: per-block partial sum / sumsq. C divides 256 so each thread
// owns a fixed channel (t % C); reduction order is fixed -> deterministic.
// ---------------------------------------------------------------------------
template <int C>
__global__ __launch_bounds__(256) void stats_partial(
    const float* __restrict__ x, float* __restrict__ partials, long long total) {
  constexpr int ITER = 64;
  const int t = threadIdx.x;
  const long long base = (long long)blockIdx.x * 256 * ITER + t;
  float s = 0.f, s2 = 0.f;
  #pragma unroll 4
  for (int i = 0; i < ITER; ++i) {
    long long idx = base + (long long)i * 256;
    if (idx < total) { float v = x[idx]; s += v; s2 += v * v; }
  }
  __shared__ float ls[256];
  __shared__ float ls2[256];
  ls[t] = s; ls2[t] = s2;
  __syncthreads();
  if (t < C) {
    float a = ls[t], b = ls2[t];
    for (int o = C; o < 256; o += C) { a += ls[t + o]; b += ls2[t + o]; }
    partials[(long long)blockIdx.x * 2 * C + t]     = a;
    partials[(long long)blockIdx.x * 2 * C + C + t] = b;
  }
}

template <int C>
__global__ __launch_bounds__(256) void stats_finalize(
    const float* __restrict__ partials, int nblk,
    const float* __restrict__ gamma, const float* __restrict__ beta,
    float* __restrict__ ss, int n) {
  const int t = threadIdx.x;
  if (t >= C) return;
  float s = 0.f, s2 = 0.f;
  for (int b = 0; b < nblk; ++b) {
    s  += partials[(long long)b * 2 * C + t];
    s2 += partials[(long long)b * 2 * C + C + t];
  }
  const float inv_n = 1.0f / (float)n;
  const float mu  = s * inv_n;
  const float var = s2 * inv_n - mu * mu;   // biased var, matches jnp.var
  const float sc  = gamma[t] * rsqrtf(var + 1e-4f);
  ss[t]     = sc;
  ss[C + t] = beta[t] - mu * sc;
}

// BN + ReLU (leak=0), convert to f16
template <int C>
__global__ __launch_bounds__(256) void bn_relu_cvt(
    const float* __restrict__ x, const float* __restrict__ ss,
    half_t* __restrict__ y, long long total) {
  long long i = (long long)blockIdx.x * 256 + threadIdx.x;
  const long long stride = (long long)gridDim.x * 256;
  for (; i < total; i += stride) {
    const int c = (int)(i % C);
    float v = x[i] * ss[c] + ss[C + c];
    v = v > 0.f ? v : 0.f;
    y[i] = (half_t)v;
  }
}

// plain f32 -> f16 convert (for raw feat used by the NiN shortcut)
__global__ __launch_bounds__(256) void cvt_f16(
    const float* __restrict__ x, half_t* __restrict__ y, long long total) {
  long long i = (long long)blockIdx.x * 256 + threadIdx.x;
  const long long stride = (long long)gridDim.x * 256;
  for (; i < total; i += stride) y[i] = (half_t)x[i];
}

// W[k][cin][cout] f32 -> Wt[k][cout][cin] f16 (B-matrix friendly layout)
__global__ __launch_bounds__(256) void transpose_cvt(
    const float* __restrict__ w, half_t* __restrict__ wt,
    int kcnt, int cin, int cout) {
  const long long total = (long long)kcnt * cin * cout;
  long long i = (long long)blockIdx.x * 256 + threadIdx.x;
  const long long stride = (long long)gridDim.x * 256;
  for (; i < total; i += stride) {
    const int k = (int)(i / (cin * cout));
    const int r = (int)(i % (cin * cout));
    const int c = r / cout;
    const int nn = r % cout;
    wt[((long long)k * cout + nn) * cin + c] = (half_t)w[i];
  }
}

// ---------------------------------------------------------------------------
// Rulebook gather-GEMM:  out[n,:] (+= addin) = sum_k gather_k(h) @ Wt[k]^T
//   h  : [N][CIN]   f16 activations
//   wt : [kcnt][128][CIN] f16 (pre-transposed weights)
//   nbr/mask: rulebook (nullptr => identity gather, mask all-true)
//   zrow: CIN halves of zeros (masked-row source)
// Block = 256 threads = 8 waves; tile = 128 rows x 128 cols.
// Each wave: 16-row strip, 8x v_wmma_f32_16x16x32_f16 accumulators.
// Weight slice for offset k staged into padded LDS with async-to-LDS DMA.
// ---------------------------------------------------------------------------
template <int CIN>
__global__ __launch_bounds__(256) void gemm_gather(
    const half_t* __restrict__ h,
    const half_t* __restrict__ wt,
    const int* __restrict__ nbr,
    const unsigned char* __restrict__ mask,
    const half_t* __restrict__ zrow,
    const float* __restrict__ addin,
    float* __restrict__ out,
    int n, int kcnt) {
  constexpr int LDW = CIN + 8;                    // pad 16B per row
  __shared__ half_t lds[128 * LDW];

  const int tid  = threadIdx.x;
  const int wave = tid >> 5;
  const int lane = tid & 31;
  const int ln   = lane & 15;                     // 0..15
  const int hi   = lane >> 4;                     // 0/1 (half-wave)

  const int rowBase = blockIdx.x * 128 + wave * 16;
  const int myRow   = rowBase + ln;
  const int rowC    = myRow < n ? myRow : (n - 1);   // clamped for safe loads

  v8f acc[8] = {};

  for (int k = 0; k < kcnt; ++k) {
    __syncthreads();                              // LDS reuse fence
    // ---- stage Wt[k] (128 x CIN halves) into padded LDS via async DMA ----
    {
      const char* src = (const char*)(wt + (long long)k * 128 * CIN);
      constexpr int CH = 128 * CIN / 8;           // 16B chunks
      #pragma unroll
      for (int c = tid; c < CH; c += 256) {
        const int row  = c / (CIN / 8);
        const int col8 = c % (CIN / 8);
        // low 32 bits of a generic shared pointer == LDS byte offset
        const unsigned ldsAddr =
            (unsigned)(size_t)&lds[row * LDW + col8 * 8];
        const unsigned voff = (unsigned)(c * 16);
        asm volatile("global_load_async_to_lds_b128 %0, %1, %2"
                     :: "v"(ldsAddr), "v"(voff), "s"(src) : "memory");
      }
      if (k + 1 < kcnt)                           // global_prefetch next slice
        __builtin_prefetch(wt + (long long)(k + 1) * 128 * CIN, 0, 2);
      asm volatile("s_wait_asynccnt 0" ::: "memory");
    }
    __syncthreads();

    // ---- rulebook gather for this offset: select source row pointer ------
    const half_t* __restrict__ arow;
    if (nbr) {
      const int  ga = nbr[(long long)rowC * KOFF + k];
      const bool ok = mask[(long long)rowC * KOFF + k] != 0;
      arow = ok ? (h + (long long)ga * CIN) : zrow;   // 1 pointer cndmask
    } else {
      arow = h + (long long)rowC * CIN;
    }

    #pragma unroll
    for (int c0 = 0; c0 < CIN; c0 += 32) {
      // A fragment 16x32 f16 (interleaved K per ISA table):
      // lane (ln,hi) holds K = c0 + {hi*8..hi*8+7, 16+hi*8..16+hi*8+7}
      V16 a;
      a.h[0] = *(const v8h*)(arow + c0 + hi * 8);
      a.h[1] = *(const v8h*)(arow + c0 + 16 + hi * 8);

      #pragma unroll
      for (int j = 0; j < 8; ++j) {               // 8 col-groups of 16
        // B fragment 32x16 f16 (interleaved K, same as A):
        // lane (ln,hi) holds column n=16j+ln, K = c0 + {hi*8..hi*8+7, 16+hi*8..16+hi*8+7}
        const half_t* b0 = &lds[(j * 16 + ln) * LDW + c0 + hi * 8];
        V16 b;
        b.h[0] = *(const v8h*)(b0);
        b.h[1] = *(const v8h*)(b0 + 16);
        acc[j] = __builtin_amdgcn_wmma_f32_16x16x32_f16(
            false, a.v, false, b.v, (short)0, acc[j], false, false);
      }
    }
  }

  // ---- store: lane (ln,hi), acc[j][r] is element (M=r+8*hi, N=16j+ln) -----
  #pragma unroll
  for (int j = 0; j < 8; ++j) {
    #pragma unroll
    for (int r = 0; r < 8; ++r) {
      const int row = rowBase + r + hi * 8;
      if (row < n) {
        const long long o = (long long)row * 128 + j * 16 + ln;
        float v = acc[j][r];
        if (addin) v += addin[o];
        out[o] = v;
      }
    }
  }
}

// ---------------------------------------------------------------------------
// Host-side launcher
// ---------------------------------------------------------------------------
extern "C" void kernel_launch(void* const* d_in, const int* in_sizes, int n_in,
                              void* d_out, int out_size, void* d_ws, size_t ws_size,
                              hipStream_t stream) {
  const float* feat   = (const float*)d_in[0];
  const float* gamma1 = (const float*)d_in[1];
  const float* beta1  = (const float*)d_in[2];
  const float* W1     = (const float*)d_in[3];
  const float* gamma2 = (const float*)d_in[4];
  const float* beta2  = (const float*)d_in[5];
  const float* W2     = (const float*)d_in[6];
  const float* Wnin   = (const float*)d_in[7];
  const int*   nbr    = (const int*)d_in[8];
  const unsigned char* mask = (const unsigned char*)d_in[9];

  const int A = 64, B = 128;
  const int n = in_sizes[0] / A;     // number of active sites

  // carve scratch from d_ws (256B aligned slabs)
  char* w = (char*)d_ws;
  size_t off = 0;
  auto alloc = [&](size_t bytes) -> void* {
    void* p = w + off;
    off = (off + bytes + 255) & ~(size_t)255;
    return p;
  };
  half_t* wt1    = (half_t*)alloc((size_t)KOFF * B * A * sizeof(half_t));
  half_t* wt2    = (half_t*)alloc((size_t)KOFF * B * B * sizeof(half_t));
  half_t* wtnin  = (half_t*)alloc((size_t)B * A * sizeof(half_t));
  half_t* zrow   = (half_t*)alloc((size_t)B * sizeof(half_t));
  half_t* feat16 = (half_t*)alloc((size_t)n * A * sizeof(half_t));
  half_t* h1     = (half_t*)alloc((size_t)n * A * sizeof(half_t));
  float*  h2     = (float*)alloc((size_t)n * B * sizeof(float));
  half_t* h2n    = (half_t*)alloc((size_t)n * B * sizeof(half_t));
  float*  ss1    = (float*)alloc(2 * A * sizeof(float));
  float*  ss2    = (float*)alloc(2 * B * sizeof(float));

  const long long totA = (long long)n * A;
  const long long totB = (long long)n * B;
  const int nStats1 = (int)((totA + 16383) / 16384);
  const int nStats2 = (int)((totB + 16383) / 16384);
  float* p1 = (float*)alloc((size_t)nStats1 * 2 * A * sizeof(float));
  float* p2 = (float*)alloc((size_t)nStats2 * 2 * B * sizeof(float));
  (void)ws_size; (void)n_in; (void)out_size;

  float* out = (float*)d_out;

  // 0) zero row for masked gathers (graph-capture-safe async memset)
  hipMemsetAsync(zrow, 0, (size_t)B * sizeof(half_t), stream);

  // 1) weight transpose + f16 convert
  {
    long long t1 = (long long)KOFF * A * B;
    long long t2 = (long long)KOFF * B * B;
    long long tn = (long long)A * B;
    transpose_cvt<<<(int)((t1 + 255) / 256), 256, 0, stream>>>(W1, wt1, KOFF, A, B);
    transpose_cvt<<<(int)((t2 + 255) / 256), 256, 0, stream>>>(W2, wt2, KOFF, B, B);
    transpose_cvt<<<(int)((tn + 255) / 256), 256, 0, stream>>>(Wnin, wtnin, 1, A, B);
  }

  // 2) feat -> f16 (NiN shortcut input)
  cvt_f16<<<2048, 256, 0, stream>>>(feat, feat16, totA);

  // 3) BN1 stats + apply -> h1 (f16)
  stats_partial<64><<<nStats1, 256, 0, stream>>>(feat, p1, totA);
  stats_finalize<64><<<1, 64, 0, stream>>>(p1, nStats1, gamma1, beta1, ss1, n);
  bn_relu_cvt<64><<<2048, 256, 0, stream>>>(feat, ss1, h1, totA);

  const int gemmGrid = (n + 127) / 128;

  // 4) y1 = feat @ Wnin  -> written into d_out (identity gather, kcnt=1)
  gemm_gather<64><<<gemmGrid, 256, 0, stream>>>(
      feat16, wtnin, nullptr, nullptr, zrow, nullptr, out, n, 1);

  // 5) h2 = subconv(h1, W1)   (27 gathered 64->128 GEMMs)
  gemm_gather<64><<<gemmGrid, 256, 0, stream>>>(
      h1, wt1, nbr, mask, zrow, nullptr, h2, n, KOFF);

  // 6) BN2 stats + apply -> h2n (f16)
  stats_partial<128><<<nStats2, 256, 0, stream>>>(h2, p2, totB);
  stats_finalize<128><<<1, 128, 0, stream>>>(p2, nStats2, gamma2, beta2, ss2, n);
  bn_relu_cvt<128><<<2048, 256, 0, stream>>>(h2, ss2, h2n, totB);

  // 7) out = y1 + subconv(h2n, W2)   (27 gathered 128->128 GEMMs, add-in y1)
  gemm_gather<128><<<gemmGrid, 256, 0, stream>>>(
      h2n, wt2, nbr, mask, zrow, out, out, n, KOFF);
}
